// _StdBlock_25769803776175
// MI455X (gfx1250) — compile-verified, matrix operand order checked by reading, not run on hardware
//
#include <hip/hip_runtime.h>

// ---------------- problem constants ----------------
constexpr int D   = 1024;
constexpr int H   = 16;
constexpr int DH  = 64;      // D / H
constexpr int DFF = 4096;    // 4 * D
constexpr int Bb  = 2;
constexpr int T   = 2048;
constexpr int M_TOK = Bb * T;          // 4096 token rows

typedef unsigned int  u32;
typedef unsigned short u16;

// ---------------- WMMA / vector types (all trivial, union-safe) ----------------
typedef __attribute__((ext_vector_type(16))) __bf16 v16bf;
typedef __attribute__((ext_vector_type(8)))  float  v8f;
typedef __attribute__((ext_vector_type(4)))  u32    v4u;

// Fragment: q[0] covers K = 8*hi .. 8*hi+7 (VGPR 0-3), q[1] covers K = 16+8*hi .. +23 (VGPR 4-7)
union Frag16 { v16bf v; u32 u[8]; v4u q[2]; };

// bf16 round-to-nearest-even
__device__ __forceinline__ u16 f2bf(float f) {
    u32 u = __float_as_uint(f);
    u32 r = 0x7FFFu + ((u >> 16) & 1u);
    return (u16)((u + r) >> 16);
}

// ---------------- elementwise cast ----------------
__global__ __launch_bounds__(256) void cast_f32_bf16_kernel(const float* __restrict__ in,
                                                            u16* __restrict__ out, int n) {
    int i = blockIdx.x * 256 + threadIdx.x;
    if (i < n) out[i] = f2bf(in[i]);
}

// ---------------- LayerNorm -> bf16 ----------------
__global__ __launch_bounds__(256) void layernorm_bf16_kernel(const float* __restrict__ x,
                                                             const float* __restrict__ g,
                                                             const float* __restrict__ bta,
                                                             u16* __restrict__ out) {
    __shared__ float red[256];
    const int row = blockIdx.x;
    const int tid = threadIdx.x;
    const float* xr = x + (size_t)row * D;
    float4 v = ((const float4*)xr)[tid];          // 256 * 4 = 1024 = D

    red[tid] = v.x + v.y + v.z + v.w; __syncthreads();
    for (int st = 128; st > 0; st >>= 1) { if (tid < st) red[tid] += red[tid + st]; __syncthreads(); }
    float mu = red[0] * (1.0f / D);
    __syncthreads();

    float d0 = v.x - mu, d1 = v.y - mu, d2 = v.z - mu, d3 = v.w - mu;
    red[tid] = d0*d0 + d1*d1 + d2*d2 + d3*d3; __syncthreads();
    for (int st = 128; st > 0; st >>= 1) { if (tid < st) red[tid] += red[tid + st]; __syncthreads(); }
    float inv = rsqrtf(red[0] * (1.0f / D) + 1e-5f);

    int c = tid * 4;
    u16* o = out + (size_t)row * D + c;
    o[0] = f2bf(d0 * inv * g[c+0] + bta[c+0]);
    o[1] = f2bf(d1 * inv * g[c+1] + bta[c+1]);
    o[2] = f2bf(d2 * inv * g[c+2] + bta[c+2]);
    o[3] = f2bf(d3 * inv * g[c+3] + bta[c+3]);
}

// ---------------- V transpose: qkv[b,t, 2D + h*DH + d] -> vt[((b*H+h)*DH+d)*T + t] ----------------
__global__ __launch_bounds__(256) void transpose_v_kernel(const u16* __restrict__ qkv,
                                                          u16* __restrict__ vt) {
    int idx = blockIdx.x * 256 + threadIdx.x;   // B*H*DH*T = 4M
    int t = idx & (T - 1);
    int r = idx >> 11;          // /T
    int d = r & (DH - 1);
    r >>= 6;                    // /DH
    int h = r & (H - 1);
    int b = r >> 4;             // /H
    vt[idx] = qkv[(size_t)(b * T + t) * (3 * D) + 2 * D + h * DH + d];
}

// ---------------- bf16 WMMA GEMM, 128x128x32 block tile, b128 fragment loads ----------------
// flags: 1=add bias[col], 2=exact GELU, 4=add resid[row*N+col] (f32), 8=store bf16 else f32
#define GF_BIAS 1
#define GF_GELU 2
#define GF_RES  4
#define GF_BF16 8

__global__ __launch_bounds__(256)
void gemm_bf16_kernel(const u16* __restrict__ A, const u16* __restrict__ Bw,
                      void* __restrict__ Cout,
                      const float* __restrict__ bias, const float* __restrict__ resid,
                      int M, int N, int K, int flags) {
    constexpr int TM = 128, TN = 128, TK = 32;
    constexpr int LDP = 40;                       // padded K-stride (80B: conflict-free, 16B aligned)
    __shared__ __align__(16) u16 sA [TM * LDP];   // m-major, k-contiguous (10 KB)
    __shared__ __align__(16) u16 sBt[TN * LDP];   // n-major, k-contiguous (transposed) (10 KB)

    const int tid  = threadIdx.x;
    const int lane = tid & 31;
    const int w    = tid >> 5;            // 8 waves
    const int waveM = w & 3;              // 4 row groups of 32
    const int waveN = w >> 2;             // 2 col groups of 64
    const int m0 = blockIdx.y * TM, n0 = blockIdx.x * TN;
    const int nlo = lane & 15, hi = lane >> 4;

    // staging coordinates
    const int ar = tid >> 1, acs = (tid & 1) * 16;      // A: 128 rows x 2 chunks of 16
    const int br = tid >> 3, bcs = (tid & 7) * 16;      // B: 32 rows x 8 chunks of 16

    v8f acc[2][4];
    const v8f vz = {0.f,0.f,0.f,0.f,0.f,0.f,0.f,0.f};
    #pragma unroll
    for (int mi = 0; mi < 2; ++mi)
        #pragma unroll
        for (int ni = 0; ni < 4; ++ni) acc[mi][ni] = vz;

    for (int kk = 0; kk < K; kk += TK) {
        const u16* gA = A  + (size_t)(m0 + ar) * K + kk + acs;
        const u16* gB = Bw + (size_t)(kk + br) * N + n0 + bcs;
        if (kk + TK < K) {                         // prefetch next K tile (global_prefetch path)
            __builtin_prefetch(gA + TK, 0, 3);
            __builtin_prefetch(gB + (size_t)TK * N, 0, 3);
        }
        // stage A tile 128x32, k-contiguous
        {
            const v4u* src = (const v4u*)gA;
            v4u v0 = src[0], v1 = src[1];
            *(v4u*)(sA + ar * LDP + acs)     = v0;
            *(v4u*)(sA + ar * LDP + acs + 8) = v1;
        }
        // stage B tile 32x128, transposed into n-major/k-contiguous
        {
            union { v4u q[2]; u16 e[16]; } t;
            const v4u* src = (const v4u*)gB;
            t.q[0] = src[0]; t.q[1] = src[1];
            #pragma unroll
            for (int j = 0; j < 16; ++j)
                sBt[(bcs + j) * LDP + br] = t.e[j];
        }
        __syncthreads();

        Frag16 af[2], bfr[4];
        #pragma unroll
        for (int mi = 0; mi < 2; ++mi) {
            const u16* p = sA + (waveM * 32 + mi * 16 + nlo) * LDP + 8 * hi;
            af[mi].q[0] = *(const v4u*)p;
            af[mi].q[1] = *(const v4u*)(p + 16);
        }
        #pragma unroll
        for (int ni = 0; ni < 4; ++ni) {
            const u16* p = sBt + (waveN * 64 + ni * 16 + nlo) * LDP + 8 * hi;
            bfr[ni].q[0] = *(const v4u*)p;
            bfr[ni].q[1] = *(const v4u*)(p + 16);
        }
        #pragma unroll
        for (int mi = 0; mi < 2; ++mi)
            #pragma unroll
            for (int ni = 0; ni < 4; ++ni)
                acc[mi][ni] = __builtin_amdgcn_wmma_f32_16x16x32_bf16(
                    false, af[mi].v, false, bfr[ni].v, (short)0, acc[mi][ni], false, false);
        __syncthreads();
    }

    // epilogue
    #pragma unroll
    for (int mi = 0; mi < 2; ++mi) {
        #pragma unroll
        for (int ni = 0; ni < 4; ++ni) {
            int col = n0 + waveN * 64 + ni * 16 + nlo;
            float bv = (flags & GF_BIAS) ? bias[col] : 0.0f;
            #pragma unroll
            for (int r = 0; r < 8; ++r) {
                int row = m0 + waveM * 32 + mi * 16 + r + 8 * hi;
                float v = acc[mi][ni][r] + bv;
                if (flags & GF_GELU) v = 0.5f * v * (1.0f + erff(v * 0.70710678118654752f));
                if (flags & GF_RES)  v += resid[(size_t)row * N + col];
                if (flags & GF_BF16) ((u16*)Cout)[(size_t)row * N + col] = f2bf(v);
                else                 ((float*)Cout)[(size_t)row * N + col] = v;
            }
        }
    }
}

// ---------------- flash attention: one wave per (b, h, 16-query tile) ----------------
__global__ __launch_bounds__(32)
void attn_kernel(const u16* __restrict__ qkv, const u16* __restrict__ vt,
                 u16* __restrict__ attn_out) {
    constexpr int PLD = 40;                       // padded row stride for P bounce
    __shared__ __align__(16) u16 psh[16 * PLD];
    const int lane = threadIdx.x;
    const int qt = blockIdx.x, h = blockIdx.y, b = blockIdx.z;
    const int t0 = qt * 16;
    const int nlo = lane & 15, hi = lane >> 4;
    const float scale = 0.125f;                   // 1/sqrt(64)
    const v8f vz = {0.f,0.f,0.f,0.f,0.f,0.f,0.f,0.f};

    // Q fragments: 16x64 A-matrix, two 16x32 K-chunks (b128 loads)
    Frag16 qf[2];
    const u16* qbase = qkv + (size_t)(b * T + t0) * (3 * D) + h * DH + (size_t)nlo * (3 * D);
    #pragma unroll
    for (int c = 0; c < 2; ++c) {
        qf[c].q[0] = *(const v4u*)(qbase + c * 32 + 8 * hi);
        qf[c].q[1] = *(const v4u*)(qbase + c * 32 + 8 * hi + 16);
    }

    v8f o[4];
    #pragma unroll
    for (int d = 0; d < 4; ++d) o[d] = vz;
    float m_i[8], l_i[8];
    #pragma unroll
    for (int r = 0; r < 8; ++r) { m_i[r] = -3.0e38f; l_i[r] = 0.0f; }

    const u16* kb0 = qkv + (size_t)b * T * (3 * D) + D + h * DH;
    const u16* vb0 = vt + (size_t)(b * H + h) * DH * T;

    for (int tk = 0; tk < T; tk += 32) {
        // ---- S = Q K^T for 32 keys (two 16-key column tiles) ----
        v8f s0 = vz, s1 = vz;
        const u16* k0p = kb0 + (size_t)(tk + nlo) * (3 * D) + 8 * hi;
        const u16* k1p = k0p + (size_t)16 * (3 * D);
        #pragma unroll
        for (int c = 0; c < 2; ++c) {
            Frag16 kf0, kf1;
            kf0.q[0] = *(const v4u*)(k0p + c * 32);
            kf0.q[1] = *(const v4u*)(k0p + c * 32 + 16);
            kf1.q[0] = *(const v4u*)(k1p + c * 32);
            kf1.q[1] = *(const v4u*)(k1p + c * 32 + 16);
            s0 = __builtin_amdgcn_wmma_f32_16x16x32_bf16(false, qf[c].v, false, kf0.v, (short)0, s0, false, false);
            s1 = __builtin_amdgcn_wmma_f32_16x16x32_bf16(false, qf[c].v, false, kf1.v, (short)0, s1, false, false);
        }

        // ---- online softmax over 32 columns ----
        float mx[8];
        #pragma unroll
        for (int r = 0; r < 8; ++r) mx[r] = fmaxf(s0[r] * scale, s1[r] * scale);
        #pragma unroll
        for (int msk = 1; msk < 16; msk <<= 1)
            #pragma unroll
            for (int r = 0; r < 8; ++r) mx[r] = fmaxf(mx[r], __shfl_xor(mx[r], msk, 32));

        float p0[8], p1[8], rs[8];
        #pragma unroll
        for (int r = 0; r < 8; ++r) {
            float nm = fmaxf(m_i[r], mx[r]);
            float alpha = __expf(m_i[r] - nm);
            m_i[r] = nm;
            p0[r] = __expf(s0[r] * scale - nm);
            p1[r] = __expf(s1[r] * scale - nm);
            rs[r] = p0[r] + p1[r];
            l_i[r] *= alpha;
            #pragma unroll
            for (int d = 0; d < 4; ++d) o[d][r] *= alpha;
        }
        #pragma unroll
        for (int msk = 1; msk < 16; msk <<= 1)
            #pragma unroll
            for (int r = 0; r < 8; ++r) rs[r] += __shfl_xor(rs[r], msk, 32);
        #pragma unroll
        for (int r = 0; r < 8; ++r) l_i[r] += rs[r];

        // ---- relayout exp(P): C layout -> 16x32 A fragment via LDS ----
        #pragma unroll
        for (int r = 0; r < 8; ++r) {
            psh[(r + 8 * hi) * PLD + nlo]      = f2bf(p0[r]);
            psh[(r + 8 * hi) * PLD + 16 + nlo] = f2bf(p1[r]);
        }
        __syncthreads();
        Frag16 pa;
        pa.q[0] = *(const v4u*)(psh + nlo * PLD + 8 * hi);
        pa.q[1] = *(const v4u*)(psh + nlo * PLD + 8 * hi + 16);
        __syncthreads();

        // ---- O += P * V (V pre-transposed: key-pairs contiguous along t) ----
        #pragma unroll
        for (int d = 0; d < 4; ++d) {
            Frag16 vf;
            const u16* vp = vb0 + (size_t)(d * 16 + nlo) * T + tk + 8 * hi;
            vf.q[0] = *(const v4u*)vp;
            vf.q[1] = *(const v4u*)(vp + 16);
            o[d] = __builtin_amdgcn_wmma_f32_16x16x32_bf16(false, pa.v, false, vf.v, (short)0, o[d], false, false);
        }
    }

    // ---- normalize and write (row-major [B*T, D], head-concatenated) ----
    #pragma unroll
    for (int d = 0; d < 4; ++d)
        #pragma unroll
        for (int r = 0; r < 8; ++r) {
            int row = t0 + r + 8 * hi;
            int col = h * DH + d * 16 + nlo;
            attn_out[(size_t)(b * T + row) * D + col] = f2bf(o[d][r] / l_i[r]);
        }
}

// ---------------- host orchestration ----------------
extern "C" void kernel_launch(void* const* d_in, const int* in_sizes, int n_in,
                              void* d_out, int out_size, void* d_ws, size_t ws_size,
                              hipStream_t stream) {
    const float* x      = (const float*)d_in[0];
    const float* ln1_g  = (const float*)d_in[1];
    const float* ln1_b  = (const float*)d_in[2];
    const float* ln2_g  = (const float*)d_in[3];
    const float* ln2_b  = (const float*)d_in[4];
    const float* w_qkv  = (const float*)d_in[5];
    const float* w_out  = (const float*)d_in[6];
    const float* w_ff1  = (const float*)d_in[7];
    const float* b_ff1  = (const float*)d_in[8];
    const float* w_ff2  = (const float*)d_in[9];
    const float* b_ff2  = (const float*)d_in[10];
    float* out = (float*)d_out;

    // workspace carve-up (256B aligned)
    char* base = (char*)d_ws;
    size_t off = 0;
    auto take = [&](size_t bytes) { char* p = base + off; off += (bytes + 255) & ~(size_t)255; return p; };
    u16*   wqkv_bf = (u16*)take((size_t)D * 3 * D * 2);
    u16*   wout_bf = (u16*)take((size_t)D * D * 2);
    u16*   wff1_bf = (u16*)take((size_t)D * DFF * 2);
    u16*   wff2_bf = (u16*)take((size_t)DFF * D * 2);
    u16*   xn1_bf  = (u16*)take((size_t)M_TOK * D * 2);
    u16*   xn2_bf  = (u16*)take((size_t)M_TOK * D * 2);
    u16*   qkv_bf  = (u16*)take((size_t)M_TOK * 3 * D * 2);
    u16*   vt_bf   = (u16*)take((size_t)Bb * H * DH * T * 2);
    u16*   attn_bf = (u16*)take((size_t)M_TOK * D * 2);
    float* x1_f    = (float*)take((size_t)M_TOK * D * 4);
    u16*   h_bf    = (u16*)take((size_t)M_TOK * DFF * 2);
    (void)ws_size; (void)in_sizes; (void)n_in; (void)out_size;

    auto cast = [&](const float* src, u16* dst, int n) {
        cast_f32_bf16_kernel<<<(n + 255) / 256, 256, 0, stream>>>(src, dst, n);
    };
    cast(w_qkv, wqkv_bf, D * 3 * D);
    cast(w_out, wout_bf, D * D);
    cast(w_ff1, wff1_bf, D * DFF);
    cast(w_ff2, wff2_bf, DFF * D);

    // 1) xn1 = LN(x)
    layernorm_bf16_kernel<<<M_TOK, 256, 0, stream>>>(x, ln1_g, ln1_b, xn1_bf);

    // 2) qkv = xn1 @ w_qkv  (bf16 out)
    gemm_bf16_kernel<<<dim3(3 * D / 128, M_TOK / 128), 256, 0, stream>>>(
        xn1_bf, wqkv_bf, qkv_bf, nullptr, nullptr, M_TOK, 3 * D, D, GF_BF16);

    // 3) V transpose for attention B-fragments
    transpose_v_kernel<<<(Bb * H * DH * T) / 256, 256, 0, stream>>>(qkv_bf, vt_bf);

    // 4) flash attention -> attn_bf
    attn_kernel<<<dim3(T / 16, H, Bb), 32, 0, stream>>>(qkv_bf, vt_bf, attn_bf);

    // 5) x1 = x + attn @ w_out  (f32 out, residual)
    gemm_bf16_kernel<<<dim3(D / 128, M_TOK / 128), 256, 0, stream>>>(
        attn_bf, wout_bf, x1_f, nullptr, x, M_TOK, D, D, GF_RES);

    // 6) xn2 = LN(x1)
    layernorm_bf16_kernel<<<M_TOK, 256, 0, stream>>>(x1_f, ln2_g, ln2_b, xn2_bf);

    // 7) h = gelu(xn2 @ w_ff1 + b_ff1)  (bf16 out)
    gemm_bf16_kernel<<<dim3(DFF / 128, M_TOK / 128), 256, 0, stream>>>(
        xn2_bf, wff1_bf, h_bf, b_ff1, nullptr, M_TOK, DFF, D, GF_BIAS | GF_GELU | GF_BF16);

    // 8) out = x1 + h @ w_ff2 + b_ff2  (f32 -> d_out)
    gemm_bf16_kernel<<<dim3(D / 128, M_TOK / 128), 256, 0, stream>>>(
        h_bf, wff2_bf, out, b_ff2, x1_f, M_TOK, D, DFF, GF_BIAS | GF_RES);
}
